// DAttentionMM_84155589198444
// MI455X (gfx1250) — matrix operand order repaired and checked by reading, not compiled, not run on hardware
//
#include <hip/hip_runtime.h>
#include <hip/hip_bf16.h>

typedef __bf16 bf16_t;
typedef __attribute__((ext_vector_type(16))) __bf16 v16bf;
typedef __attribute__((ext_vector_type(8)))  __bf16 v8bf;
typedef __attribute__((ext_vector_type(8)))  float  v8f;

#define B_    8
#define C_    256
#define H_    64
#define W_    64
#define NPIX  (B_ * H_ * W_)          /* 32768 */
#define SCALE_ 0.17677669529663689f   /* 32^-0.5 */

// ---------------------------------------------------------------- helpers

__device__ __forceinline__ v8f wmma_bf16f32(v16bf a, v16bf b, v8f c) {
  // v_wmma_f32_16x16x32_bf16  (8-arg form: neg_a, A, neg_b, B, c_mod, C, reuse_a, reuse_b)
  return __builtin_amdgcn_wmma_f32_16x16x32_bf16(false, a, false, b, (short)0, c,
                                                 false, false);
}

// A fragment 16x32 bf16, row-major [M][K], ld in elements.
// lane<16: elements 0..15 = K 0..15 ; lane>=16: K 16..31
__device__ __forceinline__ v16bf load_a_frag(const bf16_t* p, int ld, int lane) {
  int m  = lane & 15;
  int kb = (lane < 16) ? 0 : 16;
  const bf16_t* row = p + m * ld + kb;
  v16bf r;
  ((v8bf*)&r)[0] = *(const v8bf*)(row);
  ((v8bf*)&r)[1] = *(const v8bf*)(row + 8);
  return r;
}

__device__ __forceinline__ float gelu_exact(float v) {
  return 0.5f * v * (1.0f + erff(v * 0.70710678118654752f));
}

__device__ __forceinline__ float block_reduce_sum(float v, float* red, int tid) {
  red[tid] = v;
  __syncthreads();
  for (int s = 128; s > 0; s >>= 1) {
    if (tid < s) red[tid] += red[tid + s];
    __syncthreads();
  }
  float r = red[0];
  __syncthreads();
  return r;
}

// ---------------------------------------------------------------- prep kernels

// concat(x,y): NCHW f32 -> NHWC bf16 [B][64][64][512]
__global__ __launch_bounds__(256) void p1_pack(const float* __restrict__ x,
                                               const float* __restrict__ y,
                                               bf16_t* __restrict__ ximg) {
  size_t i = (size_t)blockIdx.x * 256 + threadIdx.x;
  if (i >= (size_t)NPIX * 512) return;
  int    c   = (int)(i & 511);
  size_t pix = i >> 9;
  int    b   = (int)(pix >> 12);
  int    sp  = (int)(pix & 4095);
  const float* src = (c < 256) ? x : y;
  float v = src[((size_t)b * 256 + (c & 255)) * 4096 + sp];
  ximg[i] = (bf16_t)v;
}

// fuse_w [oc][ic][3][3] f32 -> [tap][oc][ic] bf16
__global__ __launch_bounds__(256) void p2_fusew(const float* __restrict__ fw,
                                                bf16_t* __restrict__ wfuse) {
  int i = blockIdx.x * 256 + threadIdx.x;     // over 256*512*9
  if (i >= 256 * 512 * 9) return;
  int oc  = i / (512 * 9);
  int rem = i % (512 * 9);
  int ic  = rem / 9;
  int tap = rem % 9;
  wfuse[(size_t)tap * 256 * 512 + (size_t)oc * 512 + ic] = (bf16_t)fw[i];
}

// generic f32 -> bf16 copy (same layout)
__global__ __launch_bounds__(256) void p2_mat(const float* __restrict__ s,
                                              bf16_t* __restrict__ d, int n) {
  int i = blockIdx.x * 256 + threadIdx.x;
  if (i < n) d[i] = (bf16_t)s[i];
}

// fold conv bias + BN into per-channel scale/shift; also clear the zero row
__global__ __launch_bounds__(256) void p3_bnfold(const float* __restrict__ fb,
                                                 const float* __restrict__ g,
                                                 const float* __restrict__ bb,
                                                 const float* __restrict__ m,
                                                 const float* __restrict__ v,
                                                 float* __restrict__ scale,
                                                 float* __restrict__ shift,
                                                 bf16_t* __restrict__ zrow) {
  int c = threadIdx.x;
  float sc = rsqrtf(v[c] + 1e-5f) * g[c];
  scale[c] = sc;
  shift[c] = (fb[c] - m[c]) * sc + bb[c];
  zrow[c]       = (bf16_t)0.0f;
  zrow[c + 256] = (bf16_t)0.0f;
}

// ---------------------------------------------------------------- K1: conv3x3 + BN + GELU

__global__ __launch_bounds__(256) void k1_conv3x3(const bf16_t* __restrict__ ximg,
                                                  const bf16_t* __restrict__ wfuse,
                                                  const float* __restrict__ scale,
                                                  const float* __restrict__ shift,
                                                  const bf16_t* __restrict__ zrow,
                                                  bf16_t* __restrict__ hbuf) {
  int lane = threadIdx.x & 31;
  int wave = threadIdx.x >> 5;
  int ocTile = blockIdx.x * 8 + wave;       // 0..15
  int pt   = blockIdx.y;                    // 0..2047 : (b,y,xt)
  int xt   = pt & 3;
  int row  = pt >> 2;                       // b*64 + y
  int y    = row & 63;
  int b    = row >> 6;
  int xBase = xt * 16;
  int n    = lane & 15;
  int kbB  = (lane < 16) ? 0 : 16;

  v8f acc = {};
  for (int tap = 0; tap < 9; ++tap) {
    int dy = tap / 3 - 1, dx = tap % 3 - 1;
    int sy = y + dy;
    if (sy < 0 || sy > 63) continue;        // wave-uniform
    int  sx    = xBase + n + dx;
    bool valid = (sx >= 0 && sx < 64);
    // invalid lanes read a zero-filled row: uniform control flow, no exec dance
    const bf16_t* brow = valid
        ? (ximg + (((size_t)(b * 64 + sy) * 64 + sx) * 512) + kbB)
        : (zrow + kbB);
    const bf16_t* abase = wfuse + (size_t)tap * 256 * 512 + (size_t)ocTile * 16 * 512;
    for (int ic0 = 0; ic0 < 512; ic0 += 32) {
      v16bf a = load_a_frag(abase + ic0, 512, lane);
      v16bf bm;
      ((v8bf*)&bm)[0] = *(const v8bf*)(brow + ic0);
      ((v8bf*)&bm)[1] = *(const v8bf*)(brow + ic0 + 8);
      __builtin_prefetch(brow + ic0 + 128, 0, 3);
      acc = wmma_bf16f32(a, bm, acc);
    }
  }
  // epilogue: BN fold + exact GELU, store NHWC bf16 (8 contiguous channels)
  int hi     = lane >> 4;
  int ocBase = ocTile * 16 + hi * 8;
  v8bf ov;
  for (int r = 0; r < 8; ++r) {
    float v = acc[r] * scale[ocBase + r] + shift[ocBase + r];
    ov[r] = (bf16_t)gelu_exact(v);
  }
  size_t pix = (size_t)(b * 64 + y) * 64 + xBase + n;
  *(v8bf*)(hbuf + pix * 256 + ocBase) = ov;
}

// ---------------------------------------------------------------- shared 1x1 GEMM core (Cin=256)

__device__ __forceinline__ v8f gemm256_acc(const bf16_t* __restrict__ wmat,
                                           const bf16_t* __restrict__ inbuf,
                                           size_t pixBase, int ocTile, int lane) {
  v8f acc = {};
  const bf16_t* brow = inbuf + (pixBase + (lane & 15)) * 256 + ((lane < 16) ? 0 : 16);
  const bf16_t* arow = wmat + (size_t)ocTile * 16 * 256;
  for (int ic0 = 0; ic0 < 256; ic0 += 32) {
    v16bf a = load_a_frag(arow + ic0, 256, lane);
    v16bf bm;
    ((v8bf*)&bm)[0] = *(const v8bf*)(brow + ic0);
    ((v8bf*)&bm)[1] = *(const v8bf*)(brow + ic0 + 8);
    __builtin_prefetch(brow + ic0 + 64, 0, 3);
    acc = wmma_bf16f32(a, bm, acc);
  }
  return acc;
}

// K2: q = projq(h), output NHWC bf16
__global__ __launch_bounds__(256) void k2_projq(const bf16_t* __restrict__ hbuf,
                                                const bf16_t* __restrict__ wq,
                                                const float* __restrict__ bq,
                                                bf16_t* __restrict__ qbuf) {
  int lane = threadIdx.x & 31;
  int wave = threadIdx.x >> 5;
  int ocTile = blockIdx.x * 8 + wave;
  size_t pixBase = (size_t)blockIdx.y * 16;
  v8f acc = gemm256_acc(wq, hbuf, pixBase, ocTile, lane);
  int hi = lane >> 4, n = lane & 15;
  int ocBase = ocTile * 16 + hi * 8;
  v8bf ov;
  for (int r = 0; r < 8; ++r) ov[r] = (bf16_t)(acc[r] + bq[ocBase + r]);
  *(v8bf*)(qbuf + (pixBase + n) * 256 + ocBase) = ov;
}

// K6: out = projo(attn_out), output f32 NCHW -> d_out
__global__ __launch_bounds__(256) void k6_projo(const bf16_t* __restrict__ obuf,
                                                const bf16_t* __restrict__ wo,
                                                const float* __restrict__ bo,
                                                float* __restrict__ out) {
  int lane = threadIdx.x & 31;
  int wave = threadIdx.x >> 5;
  int ocTile = blockIdx.x * 8 + wave;
  size_t pixBase = (size_t)blockIdx.y * 16;
  v8f acc = gemm256_acc(wo, obuf, pixBase, ocTile, lane);
  int hi = lane >> 4, n = lane & 15;
  int ocBase = ocTile * 16 + hi * 8;
  size_t pixel = pixBase + n;
  int b   = (int)(pixel >> 12);
  int sp  = (int)(pixel & 4095);
  for (int r = 0; r < 8; ++r)
    out[((size_t)b * 256 + ocBase + r) * 4096 + sp] = acc[r] + bo[ocBase + r];
}

// ---------------------------------------------------------------- K3: offset branches

__global__ __launch_bounds__(256) void k3a_dwconv(const float* __restrict__ x,
                                                  const float* __restrict__ y,
                                                  const float* __restrict__ wx,
                                                  const float* __restrict__ bx,
                                                  const float* __restrict__ wy,
                                                  const float* __restrict__ by,
                                                  float* __restrict__ dwbuf) {
  int i = blockIdx.x * 256 + threadIdx.x;     // 2*8*64*256
  if (i >= 2 * 8 * 64 * 256) return;
  int c  = i & 255;
  int sp = (i >> 8) & 63;
  int b  = (i >> 14) & 7;
  int br = i >> 17;
  int hk = sp >> 3, wk = sp & 7;
  const float* img = br ? y : x;
  const float* w   = (br ? wy : wx) + c * 81;
  float acc = (br ? by : bx)[c];
  const float* ch = img + ((size_t)b * 256 + c) * 4096;
  for (int ky = 0; ky < 9; ++ky) {
    int sy = hk * 8 + ky - 4;
    if (sy < 0 || sy > 63) continue;
    for (int kx = 0; kx < 9; ++kx) {
      int sx = wk * 8 + kx - 4;
      if (sx < 0 || sx > 63) continue;
      acc += ch[sy * 64 + sx] * w[ky * 9 + kx];
    }
  }
  dwbuf[((size_t)(br * 8 + b) * 64 + sp) * 256 + c] = acc;
}

__global__ __launch_bounds__(256) void k3b_offsets(const float* __restrict__ dwbuf,
                                                   const float* __restrict__ lngx,
                                                   const float* __restrict__ lnbx,
                                                   const float* __restrict__ pwx,
                                                   const float* __restrict__ lngy,
                                                   const float* __restrict__ lnby,
                                                   const float* __restrict__ pwy,
                                                   float* __restrict__ gridbuf) {
  __shared__ float red[256];
  int tid = threadIdx.x;
  int blk = blockIdx.x;         // br*512 + b*64 + sp
  int sp  = blk & 63;
  int br  = blk >> 9;
  float val  = dwbuf[(size_t)blk * 256 + tid];
  float mean = block_reduce_sum(val, red, tid) * (1.0f / 256.0f);
  float d    = val - mean;
  float var  = block_reduce_sum(d * d, red, tid) * (1.0f / 256.0f);
  const float* lg = br ? lngy : lngx;
  const float* lb = br ? lnby : lnbx;
  const float* pw = br ? pwy  : pwx;
  float g  = d * rsqrtf(var + 1e-5f) * lg[tid] + lb[tid];
  float ge = gelu_exact(g);
  float off0 = block_reduce_sum(pw[tid] * ge, red, tid);
  float off1 = block_reduce_sum(pw[256 + tid] * ge, red, tid);
  if (tid == 0) {
    int hk = sp >> 3, wk = sp & 7;
    float ry = (0.5f + hk) / 7.0f * 2.0f - 1.0f;
    float rx = (0.5f + wk) / 7.0f * 2.0f - 1.0f;
    float p0 = fminf(1.0f, fmaxf(-1.0f, off0 + ry));  // y component
    float p1 = fminf(1.0f, fmaxf(-1.0f, off1 + rx));  // x component
    // grid = pos[..., ::-1]: ix from p1, iy from p0
    gridbuf[blk * 2 + 0] = (p1 + 1.0f) * 0.5f * 63.0f;
    gridbuf[blk * 2 + 1] = (p0 + 1.0f) * 0.5f * 63.0f;
  }
}

// ---------------------------------------------------------------- K4: sample + switch + projk/projv

__global__ __launch_bounds__(256) void k4_sample(const float* __restrict__ x,
                                                 const float* __restrict__ y,
                                                 const bf16_t* __restrict__ qbuf,
                                                 const float* __restrict__ gridbuf,
                                                 const float* __restrict__ sw1w,
                                                 const float* __restrict__ sw1b,
                                                 const float* __restrict__ sw2w,
                                                 const float* __restrict__ sw2b,
                                                 const float* __restrict__ wkf,
                                                 const float* __restrict__ bk,
                                                 const float* __restrict__ wvf,
                                                 const float* __restrict__ bv,
                                                 bf16_t* __restrict__ kbuf,
                                                 bf16_t* __restrict__ vbuf) {
  __shared__ float qs[256], samp[256], r1[256], red[256];
  int tid  = threadIdx.x;
  int blk  = blockIdx.x;                 // b*128 + slot
  int slot = blk & 127;
  int b    = blk >> 7;
  int gi   = (slot < 64) ? (b * 64 + slot) : (512 + b * 64 + (slot - 64));
  float ix = gridbuf[gi * 2 + 0];
  float iy = gridbuf[gi * 2 + 1];
  float x0f = floorf(ix), y0f = floorf(iy);
  float wx = ix - x0f, wy = iy - y0f;
  int x0 = min(63, max(0, (int)x0f));
  int x1 = min(63, max(0, (int)x0f + 1));
  int y0 = min(63, max(0, (int)y0f));
  int y1 = min(63, max(0, (int)y0f + 1));
  float w00 = (1 - wx) * (1 - wy), w01 = wx * (1 - wy);
  float w10 = (1 - wx) * wy,       w11 = wx * wy;
  int p00 = y0 * 64 + x0, p01 = y0 * 64 + x1, p10 = y1 * 64 + x0, p11 = y1 * 64 + x1;

  size_t cb = ((size_t)b * 256 + tid) * 4096;
  float xs = x[cb + p00] * w00 + x[cb + p01] * w01 + x[cb + p10] * w10 + x[cb + p11] * w11;
  float ys = y[cb + p00] * w00 + y[cb + p01] * w01 + y[cb + p10] * w10 + y[cb + p11] * w11;
  size_t qb = (size_t)b * 4096 * 256;
  float qsv = (float)qbuf[qb + (size_t)p00 * 256 + tid] * w00 +
              (float)qbuf[qb + (size_t)p01 * 256 + tid] * w01 +
              (float)qbuf[qb + (size_t)p10 * 256 + tid] * w10 +
              (float)qbuf[qb + (size_t)p11 * 256 + tid] * w11;
  qs[tid] = qsv;
  __syncthreads();

  float a1 = sw1b[tid];
  const float* wrow = sw1w + (size_t)tid * 256;
  for (int c = 0; c < 256; ++c) a1 += wrow[c] * qs[c];
  r1[tid] = fmaxf(a1, 0.0f);
  __syncthreads();

  float s0 = block_reduce_sum(sw2w[tid] * r1[tid], red, tid) + sw2b[0];
  float s1 = block_reduce_sum(sw2w[256 + tid] * r1[tid], red, tid) + sw2b[1];
  float mx = fmaxf(s0, s1);
  float e0 = __expf(s0 - mx), e1 = __expf(s1 - mx);
  float w0 = e0 / (e0 + e1), w1 = e1 / (e0 + e1);
  samp[tid] = w0 * xs + w1 * ys;
  __syncthreads();

  float kk = bk[tid], vv = bv[tid];
  const float* krow = wkf + (size_t)tid * 256;
  const float* vrow = wvf + (size_t)tid * 256;
  for (int c = 0; c < 256; ++c) {
    float s = samp[c];
    kk += krow[c] * s;
    vv += vrow[c] * s;
  }
  int head = tid >> 5, hc = tid & 31;
  kbuf[(((size_t)b * 8 + head) * 128 + slot) * 32 + hc] = (bf16_t)kk;   // [bh][n][hc]
  vbuf[(((size_t)b * 8 + head) * 32 + hc) * 128 + slot] = (bf16_t)vv;   // [bh][hc][n]
}

// ---------------------------------------------------------------- K5: attention

// single shared block so LDS byte offsets are known (block placed at LDS offset 0):
//   [0,       8192)  kt : k tile  [n=128][hc=32]  bf16
//   [8192,   16384)  vt : v tile  [hc=32][n=128]  bf16
//   [16384,  49152)  at : per-wave attn [m=16][n=128] bf16 x 8 waves
#define K5_KT_OFF 0
#define K5_VT_OFF 8192
#define K5_AT_OFF 16384

__global__ __launch_bounds__(256) void k5_attn(const bf16_t* __restrict__ qbuf,
                                               const bf16_t* __restrict__ kbuf,
                                               const bf16_t* __restrict__ vbuf,
                                               bf16_t* __restrict__ obuf) {
  __shared__ __align__(16) bf16_t smem[(8192 + 8192 + 32768) / 2];
  bf16_t* kt = smem;                       // [n][hc]
  bf16_t* vt = smem + K5_VT_OFF / 2;       // [hc][n]
  bf16_t* at = smem + K5_AT_OFF / 2;       // per-wave [m][n]

  int tid  = threadIdx.x;
  int lane = tid & 31;
  int wave = tid >> 5;
  int bh = blockIdx.x;        // 0..63
  int mc = blockIdx.y;        // 0..31
  int b = bh >> 3, head = bh & 7;

  // stage k/v tiles (8 KB each) into LDS with CDNA5 async copies (ASYNCcnt)
  {
    int i = tid * 16;                                  // element index, 32 B per thread per tile
    const bf16_t* kg = kbuf + (size_t)bh * 4096 + i;
    const bf16_t* vg = vbuf + (size_t)bh * 4096 + i;
    unsigned lk = (unsigned)(K5_KT_OFF + i * 2);
    unsigned lv = (unsigned)(K5_VT_OFF + i * 2);
    asm volatile("global_load_async_to_lds_b128 %0, %1, off"            :: "v"(lk),      "v"(kg)     : "memory");
    asm volatile("global_load_async_to_lds_b128 %0, %1, off offset:16"  :: "v"(lk + 16), "v"(kg)     : "memory");
    asm volatile("global_load_async_to_lds_b128 %0, %1, off"            :: "v"(lv),      "v"(vg)     : "memory");
    asm volatile("global_load_async_to_lds_b128 %0, %1, off offset:16"  :: "v"(lv + 16), "v"(vg)     : "memory");
    asm volatile("s_wait_asynccnt 0x0" ::: "memory");
  }
  __syncthreads();

  int mBase = mc * 128 + wave * 16;
  int n16 = lane & 15, hi = lane >> 4;

  // q A-fragment: lane m=n16, K = head channels
  const bf16_t* qrow = qbuf + ((size_t)b * 4096 + mBase + n16) * 256 + head * 32 + (hi ? 16 : 0);
  v16bf aq;
  ((v8bf*)&aq)[0] = *(const v8bf*)(qrow);
  ((v8bf*)&aq)[1] = *(const v8bf*)(qrow + 8);

  // logits: 8 WMMAs cover n = 0..127
  v8f acc[8];
  v8f zero = {};
  for (int t = 0; t < 8; ++t) {
    const bf16_t* kr = kt + (size_t)(t * 16 + n16) * 32 + (hi ? 16 : 0);
    v16bf bk;
    ((v8bf*)&bk)[0] = *(const v8bf*)(kr);
    ((v8bf*)&bk)[1] = *(const v8bf*)(kr + 8);
    acc[t] = wmma_bf16f32(aq, bk, zero);
  }

  // row softmax (rows live in 16-lane halves of the wave32)
  for (int r = 0; r < 8; ++r) {
    float sv[8];
    float mxv = -1e30f;
    for (int t = 0; t < 8; ++t) {
      sv[t] = acc[t][r] * SCALE_;
      mxv = fmaxf(mxv, sv[t]);
    }
    for (int msk = 1; msk < 16; msk <<= 1) mxv = fmaxf(mxv, __shfl_xor(mxv, msk, 32));
    float sum = 0.0f;
    for (int t = 0; t < 8; ++t) {
      float e = __expf(sv[t] - mxv);
      sv[t] = e;
      sum += e;
    }
    for (int msk = 1; msk < 16; msk <<= 1) sum += __shfl_xor(sum, msk, 32);
    float inv = 1.0f / sum;
    int m = r + hi * 8;
    bf16_t* arow = at + (size_t)(wave * 16 + m) * 128;
    for (int t = 0; t < 8; ++t) arow[t * 16 + n16] = (bf16_t)(sv[t] * inv);
  }
  asm volatile("s_wait_dscnt 0x0" ::: "memory");   // wave-local LDS RAW fence

  // out = attn @ v^T : M=m(16), N=hc(2x16), K=n(4x32)
  const bf16_t* arow0 = at + (size_t)wave * 16 * 128;
  for (int ct = 0; ct < 2; ++ct) {
    v8f ao = {};
    for (int kc = 0; kc < 4; ++kc) {
      const bf16_t* ar = arow0 + (size_t)n16 * 128 + kc * 32 + (hi ? 16 : 0);
      v16bf aa;
      ((v8bf*)&aa)[0] = *(const v8bf*)(ar);
      ((v8bf*)&aa)[1] = *(const v8bf*)(ar + 8);
      const bf16_t* vr = vt + (size_t)(ct * 16 + n16) * 128 + kc * 32 + (hi ? 16 : 0);
      v16bf bv2;
      ((v8bf*)&bv2)[0] = *(const v8bf*)(vr);
      ((v8bf*)&bv2)[1] = *(const v8bf*)(vr + 8);
      ao = wmma_bf16f32(aa, bv2, ao);
    }
    int c = head * 32 + ct * 16 + n16;
    for (int r = 0; r < 8; ++r) {
      int pix = mBase + r + hi * 8;
      obuf[((size_t)b * 4096 + pix) * 256 + c] = (bf16_t)ao[r];
    }
  }
}

// ---------------------------------------------------------------- launch

extern "C" void kernel_launch(void* const* d_in, const int* in_sizes, int n_in,
                              void* d_out, int out_size, void* d_ws, size_t ws_size,
                              hipStream_t stream) {
  (void)in_sizes; (void)n_in; (void)out_size; (void)ws_size;
  const float* x        = (const float*)d_in[0];
  const float* y        = (const float*)d_in[1];
  const float* fuse_w   = (const float*)d_in[2];
  const float* fuse_b   = (const float*)d_in[3];
  const float* bn_g     = (const float*)d_in[4];
  const float* bn_b     = (const float*)d_in[5];
  const float* bn_m     = (const float*)d_in[6];
  const float* bn_v     = (const float*)d_in[7];
  const float* projq_w  = (const float*)d_in[8];
  const float* projq_b  = (const float*)d_in[9];
  const float* offx_dw_w = (const float*)d_in[10];
  const float* offx_dw_b = (const float*)d_in[11];
  const float* offx_ln_g = (const float*)d_in[12];
  const float* offx_ln_b = (const float*)d_in[13];
  const float* offx_pw_w = (const float*)d_in[14];
  const float* offy_dw_w = (const float*)d_in[15];
  const float* offy_dw_b = (const float*)d_in[16];
  const float* offy_ln_g = (const float*)d_in[17];
  const float* offy_ln_b = (const float*)d_in[18];
  const float* offy_pw_w = (const float*)d_in[19];
  const float* sw1_w    = (const float*)d_in[20];
  const float* sw1_b    = (const float*)d_in[21];
  const float* sw2_w    = (const float*)d_in[22];
  const float* sw2_b    = (const float*)d_in[23];
  const float* projk_w  = (const float*)d_in[24];
  const float* projk_b  = (const float*)d_in[25];
  const float* projv_w  = (const float*)d_in[26];
  const float* projv_b  = (const float*)d_in[27];
  const float* projo_w  = (const float*)d_in[28];
  const float* projo_b  = (const float*)d_in[29];

  char* ws = (char*)d_ws;
  size_t off = 0;
  auto take = [&](size_t bytes) -> char* {
    char* p = ws + off;
    off += (bytes + 255) & ~(size_t)255;
    return p;
  };

  bf16_t* ximg   = (bf16_t*)take((size_t)NPIX * 512 * 2);   // 33.5 MB
  bf16_t* wfuse  = (bf16_t*)take((size_t)9 * 256 * 512 * 2);
  bf16_t* wq_bf  = (bf16_t*)take((size_t)256 * 256 * 2);
  bf16_t* wo_bf  = (bf16_t*)take((size_t)256 * 256 * 2);
  float*  bscale = (float*)take(256 * 4);
  float*  bshift = (float*)take(256 * 4);
  bf16_t* zrow   = (bf16_t*)take(512 * 2);                  // zero row for conv halo
  bf16_t* hbuf   = (bf16_t*)take((size_t)NPIX * 256 * 2);   // 16.8 MB
  bf16_t* qbuf   = (bf16_t*)take((size_t)NPIX * 256 * 2);   // 16.8 MB
  float*  dwbuf  = (float*)take((size_t)2 * 8 * 64 * 256 * 4);
  float*  gridb  = (float*)take((size_t)2 * 8 * 64 * 2 * 4);
  bf16_t* kbuf   = (bf16_t*)take((size_t)64 * 128 * 32 * 2);
  bf16_t* vbuf   = (bf16_t*)take((size_t)64 * 32 * 128 * 2);
  bf16_t* obuf   = (bf16_t*)take((size_t)NPIX * 256 * 2);   // 16.8 MB

  // prep
  p1_pack<<<(NPIX * 512 + 255) / 256, 256, 0, stream>>>(x, y, ximg);
  p2_fusew<<<(256 * 512 * 9 + 255) / 256, 256, 0, stream>>>(fuse_w, wfuse);
  p2_mat<<<256, 256, 0, stream>>>(projq_w, wq_bf, 256 * 256);
  p2_mat<<<256, 256, 0, stream>>>(projo_w, wo_bf, 256 * 256);
  p3_bnfold<<<1, 256, 0, stream>>>(fuse_b, bn_g, bn_b, bn_m, bn_v, bscale, bshift, zrow);

  // main pipeline
  k1_conv3x3<<<dim3(2, 2048), 256, 0, stream>>>(ximg, wfuse, bscale, bshift, zrow, hbuf);
  k2_projq<<<dim3(2, 2048), 256, 0, stream>>>(hbuf, wq_bf, projq_b, qbuf);
  k3a_dwconv<<<1024, 256, 0, stream>>>(x, y, offx_dw_w, offx_dw_b, offy_dw_w, offy_dw_b, dwbuf);
  k3b_offsets<<<1024, 256, 0, stream>>>(dwbuf, offx_ln_g, offx_ln_b, offx_pw_w,
                                        offy_ln_g, offy_ln_b, offy_pw_w, gridb);
  k4_sample<<<1024, 256, 0, stream>>>(x, y, qbuf, gridb, sw1_w, sw1_b, sw2_w, sw2_b,
                                      projk_w, projk_b, projv_w, projv_b, kbuf, vbuf);
  k5_attn<<<dim3(64, 32), 256, 0, stream>>>(qbuf, kbuf, vbuf, obuf);
  k6_projo<<<dim3(2, 2048), 256, 0, stream>>>(obuf, wo_bf, projo_b, (float*)d_out);
}
